// TimeTrajTypeGRUNet_12652973654359
// MI455X (gfx1250) — matrix operand order repaired and checked by reading, not compiled, not run on hardware
//
#include <hip/hip_runtime.h>

// ---------------- problem constants ----------------
#define BB   64
#define TT   200
#define VV   8000
#define EE   256
#define HH   512
#define G3H  1536          // 3*H
#define BT   (BB*TT)       // 12800
#define KIN  288           // padded input dim (257 -> 288, mult of 32)

typedef __attribute__((ext_vector_type(16))) __bf16 v16bf;
typedef __attribute__((ext_vector_type(8)))  float  v8f;
typedef unsigned int  u32;
typedef unsigned short u16;

union BfVec { uint4 u[2]; v16bf v; };

// ---------------- CDNA5 async global->LDS staging ----------------
#if defined(__has_builtin)
#if __has_builtin(__builtin_amdgcn_global_load_async_to_lds_b128)
#define HAS_ASYNC_LDS 1
#endif
#endif
#ifndef HAS_ASYNC_LDS
#define HAS_ASYNC_LDS 0
#endif

typedef int v4i_t __attribute__((vector_size(16)));
typedef __attribute__((address_space(1))) v4i_t gv4i_t;   // global int4
typedef __attribute__((address_space(3))) v4i_t lv4i_t;   // LDS int4

__device__ __forceinline__ void stage16(const u32* g, u32* l) {
#if HAS_ASYNC_LDS
  __builtin_amdgcn_global_load_async_to_lds_b128((gv4i_t*)g, (lv4i_t*)l, 0, 0);
#else
  *(uint4*)l = *(const uint4*)g;
#endif
}
__device__ __forceinline__ void async_wait0() {
#if HAS_ASYNC_LDS
#if __has_builtin(__builtin_amdgcn_s_wait_asynccnt)
  __builtin_amdgcn_s_wait_asynccnt(0);
#else
  asm volatile("s_wait_asynccnt 0x0" ::: "memory");
#endif
#endif
}

// f32 -> bf16 round-to-nearest-even
__device__ __forceinline__ u16 f2bf(float f) {
  u32 u = __float_as_uint(f);
  u32 r = 0x7FFFu + ((u >> 16) & 1u);
  return (u16)((u + r) >> 16);
}
__device__ __forceinline__ float sigm(float x) { return 1.f / (1.f + __expf(-x)); }

// ---- WMMA tile loads (16-bit A 16x32 / B 32x16 lane layouts, wave32) ----
// A: lanes 0-15 row M=l hold K 0..7 (v0..3) & 16..23 (v4..7); lanes 16-31 K 8..15 & 24..31
__device__ __forceinline__ v16bf load_tileA(const u32* __restrict__ row, int ki, int half) {
  BfVec r;
  r.u[0] = *(const uint4*)(row + ki * 16 + half * 4);
  r.u[1] = *(const uint4*)(row + ki * 16 + 8 + half * 4);
  return r.v;
}
// B: lanes 0-15 col N=l hold K 0..7 (v0..3) & 16..23 (v4..7); lanes 16-31 K 8..15 & 24..31
__device__ __forceinline__ v16bf load_tileB(const u32* __restrict__ row, int ki, int half) {
  BfVec r;
  r.u[0] = *(const uint4*)(row + ki * 16 + half * 4);
  r.u[1] = *(const uint4*)(row + ki * 16 + 8 + half * 4);
  return r.v;
}

// ---------------- conversion / setup kernels ----------------
__global__ void cvt_pad_bf16(const float* __restrict__ src, u16* __restrict__ dst,
                             int Ksrc, int Kdst, long total) {
  long i = (long)blockIdx.x * 256 + threadIdx.x;
  if (i >= total) return;
  long n = i / Kdst; int k = (int)(i % Kdst);
  dst[i] = (k < Ksrc) ? f2bf(src[n * (long)Ksrc + k]) : (u16)0;
}

// X[bt,0:256]=emb_loc[loc[bt]], X[bt,256]=times[bt], X[bt,257:288]=0  (bf16)
__global__ void build_x(const int* __restrict__ loc, const float* __restrict__ times,
                        const float* __restrict__ emb, u16* __restrict__ X) {
  long i = (long)blockIdx.x * 256 + threadIdx.x;
  if (i >= (long)BT * KIN) return;
  long bt = i / KIN; int c = (int)(i % KIN);
  float v = 0.f;
  if (c < EE)        v = emb[(size_t)loc[bt] * EE + c];
  else if (c == EE)  v = times[bt];
  X[i] = f2bf(v);
}

__global__ void init_h(const int* __restrict__ labels, const float* __restrict__ emb_traj,
                       int layer, float* __restrict__ h, u16* __restrict__ hbf) {
  int i = blockIdx.x * 256 + threadIdx.x;                 // 64*512
  int b = i >> 9, j = i & 511;
  float v = emb_traj[(size_t)labels[b] * (2 * HH) + layer * HH + j];
  h[i] = v; hbf[i] = f2bf(v);
}

__global__ void relu_cvt(const float* __restrict__ src, u16* __restrict__ dst, long total) {
  long i = (long)blockIdx.x * 256 + threadIdx.x;
  if (i >= total) return;
  float v = src[i];
  dst[i] = f2bf(v > 0.f ? v : 0.f);
}

// ---------------- bf16 WMMA GEMM:  C[M,ldc] = A[M,K] * B[N,K]^T + bias ----------------
// Block = one 16-row A tile, async-staged to LDS and shared by 8 waves.
// Each wave computes a 16(M) x 64(N) tile: 4 f32 accumulators, pipelined K loop.
__global__ void gemm_bf16_wmma(const u32* __restrict__ A, const u32* __restrict__ Bm,
                               const float* __restrict__ bias, float* __restrict__ C,
                               int Nt64, int kiters, int Kpairs, int ldc) {
  __shared__ __align__(16) u32 As[16 * 256];              // up to 16 x 512 bf16
  int m_tile = blockIdx.x;
  int w = threadIdx.x >> 5;
  int lane = threadIdx.x & 31;
  int half = lane >> 4, lrow = lane & 15;

  // stage contiguous 16-row A block into LDS (ASYNCcnt path)
  const u32* Ab = A + (size_t)m_tile * 16 * Kpairs;
  int nv4 = (16 * Kpairs) >> 2;
  for (int i = threadIdx.x; i < nv4; i += 256) stage16(Ab + i * 4, &As[i * 4]);
  async_wait0();
  __syncthreads();

  int n4 = blockIdx.y * 8 + w;
  if (n4 >= Nt64) return;                                 // wave-uniform; no barriers after

  const u32* arow = &As[lrow * Kpairs];
  const u32* brow[4];
  int ncol[4];
#pragma unroll
  for (int j = 0; j < 4; ++j) {
    ncol[j] = n4 * 64 + j * 16 + lrow;
    brow[j] = Bm + (size_t)ncol[j] * Kpairs;
  }
  v8f acc[4];
#pragma unroll
  for (int j = 0; j < 4; ++j)
#pragma unroll
    for (int e = 0; e < 8; ++e) acc[j][e] = 0.f;

  // software-pipelined K loop: fetch ki+1 while WMMAs consume ki
  v16bf av = load_tileA(arow, 0, half);
  v16bf bv[4];
#pragma unroll
  for (int j = 0; j < 4; ++j) bv[j] = load_tileB(brow[j], 0, half);

  for (int ki = 0; ki < kiters; ++ki) {
    int kn = (ki + 1 < kiters) ? ki + 1 : ki;             // last iter reloads (harmless)
    v16bf av2 = load_tileA(arow, kn, half);
    v16bf bv2[4];
#pragma unroll
    for (int j = 0; j < 4; ++j) bv2[j] = load_tileB(brow[j], kn, half);
#pragma unroll
    for (int j = 0; j < 4; ++j)
      acc[j] = __builtin_amdgcn_wmma_f32_16x16x32_bf16(false, av, false, bv[j],
                                                       (short)0, acc[j], false, false);
    av = av2;
#pragma unroll
    for (int j = 0; j < 4; ++j) bv[j] = bv2[j];
  }

  int rbase = m_tile * 16 + half * 8;
#pragma unroll
  for (int j = 0; j < 4; ++j) {
    float bs = bias[ncol[j]];
#pragma unroll
    for (int vi = 0; vi < 8; ++vi)
      C[(size_t)(rbase + vi) * ldc + ncol[j]] = acc[j][vi] + bs;
  }
}

// ---------------- fused GRU step: gh = h*Whh^T ; gates ; h update ----------------
// 16 blocks x 8 waves; all waves in a block share one 16-row h tile (LDS, async-staged).
// Each wave owns hidden-column tile jt: r/z/n tiles at cols j, j+H, j+2H fuse in epilogue.
__global__ void gru_step(const u32* __restrict__ hbf_cur, const float* __restrict__ h_cur,
                         const u32* __restrict__ Whh, const float* __restrict__ bhh,
                         const float* __restrict__ gx, int t,
                         float* __restrict__ h_next, u16* __restrict__ hbf_next,
                         float* __restrict__ seq_f, u16* __restrict__ seq_b) {
  __shared__ __align__(16) u32 As[16 * (HH / 2)];         // 16 x 512 bf16 = 16KB
  int w = threadIdx.x >> 5;
  int gw = blockIdx.x * 8 + w;                            // 128 waves: 4 Mtiles x 32 jt
  int lane = threadIdx.x & 31;
  int half = lane >> 4, lrow = lane & 15;
  int m_tile = gw >> 5;                                   // uniform within block
  int jt     = gw & 31;
  int jn     = jt * 16 + lrow;

  const u32* Ab = hbf_cur + (size_t)m_tile * 16 * (HH / 2);
  for (int i = threadIdx.x; i < (16 * (HH / 2)) / 4; i += 256) stage16(Ab + i * 4, &As[i * 4]);
  async_wait0();
  __syncthreads();

  const u32* arow  = &As[lrow * (HH / 2)];
  const u32* browr = Whh + (size_t)jn * (HH / 2);
  const u32* browz = Whh + (size_t)(HH  + jn) * (HH / 2);
  const u32* brown = Whh + (size_t)(2 * HH + jn) * (HH / 2);

  v8f ar, az, an;
#pragma unroll
  for (int e = 0; e < 8; ++e) { ar[e] = 0.f; az[e] = 0.f; an[e] = 0.f; }

  v16bf av = load_tileA(arow, 0, half);
  v16bf br = load_tileB(browr, 0, half);
  v16bf bz = load_tileB(browz, 0, half);
  v16bf bn = load_tileB(brown, 0, half);
  for (int ki = 0; ki < HH / 32; ++ki) {
    int kn = (ki + 1 < HH / 32) ? ki + 1 : ki;
    v16bf av2 = load_tileA(arow, kn, half);
    v16bf br2 = load_tileB(browr, kn, half);
    v16bf bz2 = load_tileB(browz, kn, half);
    v16bf bn2 = load_tileB(brown, kn, half);
    ar = __builtin_amdgcn_wmma_f32_16x16x32_bf16(false, av, false, br, (short)0, ar, false, false);
    az = __builtin_amdgcn_wmma_f32_16x16x32_bf16(false, av, false, bz, (short)0, az, false, false);
    an = __builtin_amdgcn_wmma_f32_16x16x32_bf16(false, av, false, bn, (short)0, an, false, false);
    av = av2; br = br2; bz = bz2; bn = bn2;
  }

  float br_ = bhh[jn], bz_ = bhh[HH + jn], bn_ = bhh[2 * HH + jn];
  int rbase = m_tile * 16 + half * 8;
#pragma unroll
  for (int vi = 0; vi < 8; ++vi) {
    int b = rbase + vi;
    size_t bt = (size_t)b * TT + t;
    const float* g = gx + bt * G3H;
    float xr = g[jn], xz = g[HH + jn], xn = g[2 * HH + jn];
    float r = sigm(xr + ar[vi] + br_);
    float z = sigm(xz + az[vi] + bz_);
    float n = tanhf(xn + r * (an[vi] + bn_));
    float hp = h_cur[(size_t)b * HH + jn];
    float hn = (1.f - z) * n + z * hp;
    h_next[(size_t)b * HH + jn]   = hn;
    hbf_next[(size_t)b * HH + jn] = f2bf(hn);
    seq_f[bt * HH + jn] = hn;
    seq_b[bt * HH + jn] = f2bf(hn);
  }
}

// ---------------- epilogue heads ----------------
// Online (max, sum) in one read pass, then one read-modify-write pass.
__global__ void log_softmax_rows(float* __restrict__ out) {
  __shared__ float redm[256];
  __shared__ float reds[256];
  float* p = out + (size_t)blockIdx.x * VV;
  int tid = threadIdx.x;
  float m = -1e30f, s = 0.f;
  for (int i = tid; i < VV; i += 256) {
    float x = p[i];
    float mn = fmaxf(m, x);
    s = s * __expf(m - mn) + __expf(x - mn);
    m = mn;
  }
  redm[tid] = m; reds[tid] = s; __syncthreads();
  for (int d = 128; d > 0; d >>= 1) {
    if (tid < d) {
      float m2 = redm[tid + d], s2 = reds[tid + d];
      float mn = fmaxf(redm[tid], m2);
      reds[tid] = reds[tid] * __expf(redm[tid] - mn) + s2 * __expf(m2 - mn);
      redm[tid] = mn;
    }
    __syncthreads();
  }
  float lse = redm[0] + __logf(reds[0]);
  for (int i = tid; i < VV; i += 256) p[i] -= lse;
}

__global__ void time_head(const float* __restrict__ out2, const float* __restrict__ fcw,
                          const float* __restrict__ fcb, float* __restrict__ dst) {
  int wid = (blockIdx.x * blockDim.x + threadIdx.x) >> 5;
  int lane = threadIdx.x & 31;
  if (wid >= BT) return;
  const float* row = out2 + (size_t)wid * HH;
  const float* w   = fcw + (size_t)VV * HH;               // last fc row
  float s = 0.f;
  for (int k = lane; k < HH; k += 32) { float v = row[k]; s += (v > 0.f ? v : 0.f) * w[k]; }
  for (int off = 16; off > 0; off >>= 1) s += __shfl_down(s, off, 32);
  if (lane == 0) dst[wid] = sigm(s + fcb[VV]);
}

// ---------------- workspace layout (bytes) ----------------
static const size_t OFF_WIH0 = 0;
static const size_t OFF_WHH0 = OFF_WIH0 + (size_t)G3H * KIN * 2;
static const size_t OFF_WIH1 = OFF_WHH0 + (size_t)G3H * HH * 2;
static const size_t OFF_WHH1 = OFF_WIH1 + (size_t)G3H * HH * 2;
static const size_t OFF_FCW  = OFF_WHH1 + (size_t)G3H * HH * 2;
static const size_t OFF_X    = OFF_FCW  + (size_t)VV * HH * 2;
static const size_t OFF_GX   = OFF_X    + (size_t)BT * KIN * 2;
static const size_t OFF_SEQF = OFF_GX   + (size_t)BT * G3H * 4;
static const size_t OFF_SEQB = OFF_SEQF + (size_t)BT * HH * 4;
static const size_t OFF_HA   = OFF_SEQB + (size_t)BT * HH * 2;
static const size_t OFF_HB   = OFF_HA   + (size_t)BB * HH * 4;
static const size_t OFF_HAB  = OFF_HB   + (size_t)BB * HH * 4;
static const size_t OFF_HBB  = OFF_HAB  + (size_t)BB * HH * 2;

extern "C" void kernel_launch(void* const* d_in, const int* in_sizes, int n_in,
                              void* d_out, int out_size, void* d_ws, size_t ws_size,
                              hipStream_t stream) {
  (void)in_sizes; (void)n_in; (void)out_size; (void)ws_size;
  const int*   loc      = (const int*)  d_in[0];
  const float* times    = (const float*)d_in[1];
  const int*   labels   = (const int*)  d_in[2];
  const float* emb_loc  = (const float*)d_in[3];
  const float* emb_traj = (const float*)d_in[4];
  const float* w_ih0 = (const float*)d_in[5];
  const float* w_hh0 = (const float*)d_in[6];
  const float* b_ih0 = (const float*)d_in[7];
  const float* b_hh0 = (const float*)d_in[8];
  const float* w_ih1 = (const float*)d_in[9];
  const float* w_hh1 = (const float*)d_in[10];
  const float* b_ih1 = (const float*)d_in[11];
  const float* b_hh1 = (const float*)d_in[12];
  const float* fc_w  = (const float*)d_in[13];
  const float* fc_b  = (const float*)d_in[14];

  char* ws = (char*)d_ws;
  u16* wih0b = (u16*)(ws + OFF_WIH0);
  u16* whh0b = (u16*)(ws + OFF_WHH0);
  u16* wih1b = (u16*)(ws + OFF_WIH1);
  u16* whh1b = (u16*)(ws + OFF_WHH1);
  u16* fcwb  = (u16*)(ws + OFF_FCW);
  u16* Xbf   = (u16*)(ws + OFF_X);
  float* gx  = (float*)(ws + OFF_GX);
  float* seqf = (float*)(ws + OFF_SEQF);
  u16*   seqb = (u16*)  (ws + OFF_SEQB);
  float* hA  = (float*)(ws + OFF_HA);
  float* hB  = (float*)(ws + OFF_HB);
  u16*   hAb = (u16*)  (ws + OFF_HAB);
  u16*   hBb = (u16*)  (ws + OFF_HBB);
  float* out = (float*)d_out;

  auto blocks = [](long total) { return (int)((total + 255) / 256); };

  // 1) one-time bf16 weight conversion (padded where needed)
  cvt_pad_bf16<<<blocks((long)G3H * KIN), 256, 0, stream>>>(w_ih0, wih0b, EE + 1, KIN, (long)G3H * KIN);
  cvt_pad_bf16<<<blocks((long)G3H * HH),  256, 0, stream>>>(w_hh0, whh0b, HH, HH, (long)G3H * HH);
  cvt_pad_bf16<<<blocks((long)G3H * HH),  256, 0, stream>>>(w_ih1, wih1b, HH, HH, (long)G3H * HH);
  cvt_pad_bf16<<<blocks((long)G3H * HH),  256, 0, stream>>>(w_hh1, whh1b, HH, HH, (long)G3H * HH);
  cvt_pad_bf16<<<blocks((long)VV * HH),   256, 0, stream>>>(fc_w,  fcwb,  HH, HH, (long)VV * HH);

  // 2) fused gather + concat(time) + pad -> bf16 X
  build_x<<<blocks((long)BT * KIN), 256, 0, stream>>>(loc, times, emb_loc, Xbf);

  // 3) gx0 = X * Wih0^T + b_ih0   [12800 x 1536]
  {
    dim3 g(BT / 16, (24 + 7) / 8);
    gemm_bf16_wmma<<<g, 256, 0, stream>>>((const u32*)Xbf, (const u32*)wih0b,
                                          b_ih0, gx, 24, KIN / 32, KIN / 2, G3H);
  }

  // 4) layer-1 scan (double-buffered hidden state)
  init_h<<<(BB * HH) / 256, 256, 0, stream>>>(labels, emb_traj, 0, hA, hAb);
  for (int t = 0; t < TT; ++t) {
    float* hc = (t & 1) ? hB : hA;   u16* hcb = (t & 1) ? hBb : hAb;
    float* hn = (t & 1) ? hA : hB;   u16* hnb = (t & 1) ? hAb : hBb;
    gru_step<<<16, 256, 0, stream>>>((const u32*)hcb, hc, (const u32*)whh0b, b_hh0,
                                     gx, t, hn, hnb, seqf, seqb);
  }

  // 5) gx1 = h1seq(bf16) * Wih1^T + b_ih1  (overwrites gx)
  {
    dim3 g(BT / 16, (24 + 7) / 8);
    gemm_bf16_wmma<<<g, 256, 0, stream>>>((const u32*)seqb, (const u32*)wih1b,
                                          b_ih1, gx, 24, HH / 32, HH / 2, G3H);
  }

  // 6) layer-2 scan (seqf becomes GRU output sequence)
  init_h<<<(BB * HH) / 256, 256, 0, stream>>>(labels, emb_traj, 1, hA, hAb);
  for (int t = 0; t < TT; ++t) {
    float* hc = (t & 1) ? hB : hA;   u16* hcb = (t & 1) ? hBb : hAb;
    float* hn = (t & 1) ? hA : hB;   u16* hnb = (t & 1) ? hAb : hBb;
    gru_step<<<16, 256, 0, stream>>>((const u32*)hcb, hc, (const u32*)whh1b, b_hh1,
                                     gx, t, hn, hnb, seqf, seqb);
  }

  // 7) relu(out) -> bf16 for FC
  relu_cvt<<<blocks((long)BT * HH), 256, 0, stream>>>(seqf, seqb, (long)BT * HH);

  // 8) logits[:, :8000] = relu(out)*fc_w^T + fc_b  -> straight into d_out
  {
    dim3 g(BT / 16, (125 + 7) / 8);
    gemm_bf16_wmma<<<g, 256, 0, stream>>>((const u32*)seqb, (const u32*)fcwb,
                                          fc_b, out, 125, HH / 32, HH / 2, VV);
  }

  // 9) in-place log-softmax over 8000 classes per (b,t)
  log_softmax_rows<<<BT, 256, 0, stream>>>(out);

  // 10) time head: sigmoid(relu(out) . fc_w[8000] + fc_b[8000])
  time_head<<<(BT / 8), 256, 0, stream>>>(seqf, fc_w, fc_b, out + (size_t)BT * VV);
}
